// GMIC_71691594105373
// MI455X (gfx1250) — matrix operand order repaired and checked by reading, not compiled, not run on hardware
//
#include <hip/hip_runtime.h>
#include <hip/hip_bf16.h>
#include <cstdint>
#include <cstddef>

typedef __attribute__((ext_vector_type(16))) _Float16 v16h;
typedef __attribute__((ext_vector_type(8)))  _Float16 v8h;
typedef __attribute__((ext_vector_type(8)))  float    v8f;

#define N_IMG 4
#define GH 2944
#define GW 1920
#define CAMH 46
#define CAMW 30
#define CAMHW (CAMH*CAMW)   // 1380
#define CROP 256
#define KCROPS 6

__device__ __forceinline__ float sigmoidf_(float x){ return 1.0f/(1.0f+__expf(-x)); }

union FragH { v16h v; v8h q[2]; _Float16 h[16]; };
union FragF { v8f  v; float    f[8];  };

// ---------------------------------------------------------------- cvt f32->f16
__global__ void cvt_f32_f16(const float* __restrict__ src, _Float16* __restrict__ dst, int n){
  for (int i = blockIdx.x*blockDim.x + threadIdx.x; i < n; i += gridDim.x*blockDim.x)
    dst[i] = (_Float16)src[i];
}

// -------------------------------------- repack conv weight [CO][CI][3][3] f32
// ->  [co][tap][ci] f16  (tap = ky*3+kx), so A-fragment loads are contiguous.
__global__ void repack_w3(const float* __restrict__ src, _Float16* __restrict__ dst,
                          int CO, int CI){
  const int total = CO*CI*9;
  for (int i = blockIdx.x*blockDim.x + threadIdx.x; i < total; i += gridDim.x*blockDim.x){
    int co = i / (CI*9); int r = i - co*CI*9;
    int tap = r / CI;    int ci = r - tap*CI;
    dst[i] = (_Float16)src[(co*CI + ci)*9 + tap];
  }
}

// ===================================================================
// 3x3 stride-2 conv, NHWC, tap-major implicit GEMM, MT x 2 register blocking.
// Wave computes (MT*16)(M=co) x 32(N=spatial); K = ci in chunks of 32 per tap.
// Activations NHWC: act[(img*H*W + y*W + x)*C + c].  Weights [co][9][ci].
// CO multiple of MT*16.  All loads unconditional (clamped) + cndmask select.
// ===================================================================
template<int CIT, int MT>
__global__ void conv3_wmma_blk(const _Float16* __restrict__ in, const _Float16* __restrict__ wp,
                               _Float16* __restrict__ outh,
                               int NB, int HI, int WI, int CO, int HO, int WO)
{
  const int HOWO = HO*WO;
  const int NSP  = NB*HOWO;
  const int HIWI = HI*WI;
  const int tilesM = CO / (MT*16);
  const int tilesN = (NSP + 31) >> 5;
  const int tile = blockIdx.x*(blockDim.x >> 5) + (threadIdx.x >> 5);
  if (tile >= tilesM*tilesN) return;          // wave-uniform exit
  const int tm = tile % tilesM;
  const int tn = tile / tilesM;
  const int lane = threadIdx.x & 31;
  const int l15  = lane & 15;
  const int sub  = lane >> 4;

  // --- two N subtiles: spatial decode (once per wave) ---
  int  sp[2], ibase[2], oyS[2], oxS[2];
  bool spv[2];
  #pragma unroll
  for (int t = 0; t < 2; ++t) {
    sp[t]  = tn*32 + t*16 + l15;
    spv[t] = sp[t] < NSP;
    int s_ = spv[t] ? sp[t] : 0;
    int im = s_ / HOWO, rm = s_ - (s_/HOWO)*HOWO;
    int oy = rm / WO,   ox = rm - (rm/WO)*WO;
    oyS[t] = oy*2; oxS[t] = ox*2;
    ibase[t] = im*HIWI;
  }

  // --- MT M subtiles: weight row pointers ([co][9][CIT]) ---
  const _Float16* wrm[MT];
  #pragma unroll
  for (int mi = 0; mi < MT; ++mi)
    wrm[mi] = wp + (size_t)(tm*(MT*16) + mi*16 + l15)*9*CIT;

  FragF C[MT][2];
  #pragma unroll
  for (int mi = 0; mi < MT; ++mi)
    #pragma unroll
    for (int ni = 0; ni < 2; ++ni)
      #pragma unroll
      for (int r = 0; r < 8; ++r) C[mi][ni].f[r] = 0.f;

  const v8h z8 = {};

  for (int ci0 = 0; ci0 < CIT; ci0 += 32) {
    #pragma unroll 3
    for (int tap = 0; tap < 9; ++tap) {
      const int ky = tap / 3, kx = tap - (tap/3)*3;
      FragH A[MT], B[2];
      // ---- A: contiguous [tap][ci] runs; halves 0..7 -> ci0+8*sub, 8..15 -> +16
      #pragma unroll
      for (int mi = 0; mi < MT; ++mi) {
        const _Float16* pa = wrm[mi] + tap*CIT + ci0 + sub*8;
        A[mi].q[0] = *(const v8h*)pa;
        if constexpr (CIT >= 32) {
          A[mi].q[1] = *(const v8h*)(pa + 16);
        } else {
          A[mi].q[1] = z8;                 // ci >= CIT
        }
      }
      // ---- B: halves 0..7 -> pixel*CIT + ci0 + 8*sub, 8..15 -> +16
      #pragma unroll
      for (int t = 0; t < 2; ++t) {
        int iy = oyS[t] + ky, ix = oxS[t] + kx;
        bool vld = spv[t] && iy < HI && ix < WI;
        int iyc = iy < HI ? iy : HI-1;
        int ixc = ix < WI ? ix : WI-1;
        if constexpr (CIT >= 32) {
          const v8h* pb = (const v8h*)(in + ((size_t)ibase[t] + (size_t)iyc*WI + ixc)*CIT
                                          + ci0 + sub*8);
          v8h b0 = pb[0];
          v8h b1 = pb[2];
          B[t].q[0] = vld ? b0 : z8;
          B[t].q[1] = vld ? b1 : z8;
        } else {                           // CIT == 16: sub covers ci 8*sub..8*sub+7
          bool vv = vld;
          const v8h* pb = (const v8h*)(in + ((size_t)ibase[t] + (size_t)iyc*WI + ixc)*CIT + sub*8);
          const v8h* pp = vv ? pb : (const v8h*)in;   // address select, no branch
          v8h b0 = pp[0];
          v8h b1 = pp[1];
          B[t].q[0] = vv ? b0 : z8;
          B[t].q[1] = z8;
        }
      }
      #pragma unroll
      for (int mi = 0; mi < MT; ++mi) {
        C[mi][0].v = __builtin_amdgcn_wmma_f32_16x16x32_f16(false, A[mi].v, false, B[0].v,
                                                            (short)0, C[mi][0].v, false, false);
        C[mi][1].v = __builtin_amdgcn_wmma_f32_16x16x32_f16(false, A[mi].v, false, B[1].v,
                                                            (short)0, C[mi][1].v, false, false);
      }
    }
  }

  // ---- store NHWC: 8 consecutive halves per accumulator per lane ----
  #pragma unroll
  for (int ni = 0; ni < 2; ++ni) {
    if (!spv[ni]) continue;
    #pragma unroll
    for (int mi = 0; mi < MT; ++mi) {
      v8h o;
      #pragma unroll
      for (int r = 0; r < 8; ++r) {
        float v = C[mi][ni].f[r];
        v = v > 0.f ? v : 0.f;             // ReLU (all blocked layers)
        o[r] = (_Float16)v;
      }
      *(v8h*)(outh + (size_t)sp[ni]*CO + tm*(MT*16) + mi*16 + sub*8) = o;
    }
  }
}

// ===================================================================
// Generic packed-K implicit-GEMM conv, NHWC (CI==1 layers and 1x1 convs).
// Clamp-address + select: no predicated loads.
// ===================================================================
template<int KS, int S>
__global__ void conv_wmma(const _Float16* __restrict__ in, const _Float16* __restrict__ w,
                          _Float16* __restrict__ outh, float* __restrict__ outf,
                          int NB, int CI, int HI, int WI, int CO, int HO, int WO, int act)
{
  const int KK    = CI*KS*KS;
  const int HOWO  = HO*WO;
  const int NSP   = NB*HOWO;
  const int tilesM = (CO + 15) >> 4;
  const int tilesN = (NSP + 15) >> 4;
  const int tile = blockIdx.x*(blockDim.x >> 5) + (threadIdx.x >> 5);
  if (tile >= tilesM*tilesN) return;
  const int tm = tile % tilesM;
  const int tn = tile / tilesM;
  const int lane = threadIdx.x & 31;
  const int l15  = lane & 15;
  const int sub  = lane >> 4;

  const int sp  = tn*16 + l15;
  const bool spv = sp < NSP;
  int s_ = spv ? sp : 0;
  int img = s_ / HOWO, rem = s_ - (s_/HOWO)*HOWO;
  int oy = rem / WO,   ox = rem - (rem/WO)*WO;
  const int oyS = oy*S, oxS = ox*S;
  const int ibase = img*HI*WI;

  const int mrowA = tm*16 + l15;
  const _Float16* __restrict__ wrow = w + (size_t)(mrowA < CO ? mrowA : 0)*KK;

  FragH A, B;
  FragF C;
  #pragma unroll
  for (int r = 0; r < 8; ++r) C.f[r] = 0.f;

  for (int k0 = 0; k0 < KK; k0 += 32) {
    #pragma unroll
    for (int e = 0; e < 16; ++e) {
      // A packed-K layout [co][KK]
      int ka  = k0 + (e & 7) + ((e >> 3) << 4) + (sub << 3);
      bool av = (mrowA < CO) && (ka < KK);
      int kac = ka < KK ? ka : KK-1;
      _Float16 a = wrow[kac];
      A.h[e] = av ? a : (_Float16)0.f;
      // B: NHWC gather
      int kb  = k0 + (e & 7) + ((e >> 3) << 4) + (sub << 3);
      int kbc = kb < KK ? kb : KK-1;
      int ci  = kbc / (KS*KS);
      int rr  = kbc - ci*(KS*KS);
      int ky  = rr / KS, kx = rr - (rr/KS)*KS;
      int iy = oyS + ky, ix = oxS + kx;
      bool vld = spv && (kb < KK) && iy < HI && ix < WI;
      int iyc = iy < HI ? iy : HI-1;
      int ixc = ix < WI ? ix : WI-1;
      const _Float16* pb = in + ((size_t)ibase + (size_t)iyc*WI + ixc)*CI + ci;
      _Float16 b = *pb;
      B.h[e] = vld ? b : (_Float16)0.f;
    }
    C.v = __builtin_amdgcn_wmma_f32_16x16x32_f16(false, A.v, false, B.v,
                                                 (short)0, C.v, false, false);
  }

  if (spv) {
    #pragma unroll
    for (int r = 0; r < 8; ++r) {
      int mg = tm*16 + r + (sub << 3);
      if (mg < CO) {
        float v = C.f[r];
        if (act == 1)      v = v > 0.f ? v : 0.f;
        else if (act == 2) v = sigmoidf_(v);
        size_t oi = (size_t)sp*CO + mg;            // NHWC
        if (outh) outh[oi] = (_Float16)v;
        if (outf) outf[oi] = v;
      }
    }
  }
}

// ------------------------------------------------------------------ ROI argmax
// saliency NHWC: sal[(n*1380 + pix)*3 + c]
__global__ void roi_kernel(const float* __restrict__ sal, int* __restrict__ pos){
  __shared__ float m[CAMHW];
  __shared__ float bv[256];
  __shared__ int   bi[256];
  const int n = blockIdx.x, tid = threadIdx.x;
  for (int i = tid; i < CAMHW; i += blockDim.x) {
    const float* p = sal + ((size_t)n*CAMHW + i)*3;
    m[i] = p[0] + p[1] + p[2];
  }
  __syncthreads();
  const int PH = CAMH - 3, PW = CAMW - 3;   // 43 x 27 valid window positions
  for (int it = 0; it < KCROPS; ++it) {
    float best = -1e30f; int bidx = 0x7fffffff;
    for (int p = tid; p < PH*PW; p += blockDim.x) {
      int px = p / PW, py = p - px*PW;
      float s = 0.f;
      for (int dy = 0; dy < 4; ++dy)
        for (int dx = 0; dx < 4; ++dx)
          s += m[(px+dy)*CAMW + py + dx];
      if (s > best || (s == best && p < bidx)) { best = s; bidx = p; }
    }
    bv[tid] = best; bi[tid] = bidx; __syncthreads();
    for (int off = 128; off; off >>= 1) {
      if (tid < off) {
        if (bv[tid+off] > bv[tid] || (bv[tid+off] == bv[tid] && bi[tid+off] < bi[tid])) {
          bv[tid] = bv[tid+off]; bi[tid] = bi[tid+off];
        }
      }
      __syncthreads();
    }
    if (tid == 0) {
      int px = bi[0] / PW, py = bi[0] - (bi[0]/PW)*PW;
      int a = px * 64; if (a > GH - CROP) a = GH - CROP;   // 2944/46 == 64 exactly
      int b = py * 64; if (b > GW - CROP) b = GW - CROP;   // 1920/30 == 64 exactly
      pos[(n*KCROPS + it)*2 + 0] = a;
      pos[(n*KCROPS + it)*2 + 1] = b;
      for (int dy = 0; dy < 4; ++dy)
        for (int dx = 0; dx < 4; ++dx)
          m[(px+dy)*CAMW + py + dx] = 0.f;
    }
    __syncthreads();
  }
}

// ------------------------------------------------------------------ crop + cvt
__global__ void crops_kernel(const float* __restrict__ x, const int* __restrict__ pos,
                             _Float16* __restrict__ crops){
  const int c = blockIdx.y;                            // 0..23
  const int idx = blockIdx.x*blockDim.x + threadIdx.x; // 0..65535
  if (idx >= CROP*CROP) return;
  const int img = c / KCROPS;
  const int a = pos[c*2], b = pos[c*2 + 1];
  const int r = idx >> 8, col = idx & 255;
  crops[(size_t)c*CROP*CROP + idx] =
      (_Float16)x[((size_t)img*GH + (a + r))*GW + (b + col)];
}

// ----------------------------------------------------- global avg pool (NHWC)
__global__ void gap_kernel(const _Float16* __restrict__ in, float* __restrict__ out,
                           int hw){
  const int i = blockIdx.x*blockDim.x + threadIdx.x;   // (crop*512 + ch)
  if (i >= 24*512) return;
  const int crop = i >> 9, ch = i & 511;
  float s = 0.f;
  for (int j = 0; j < hw; ++j) s += (float)in[((size_t)crop*hw + j)*512 + ch];
  out[i] = s / (float)hw;
}

// --------------------------------------------------------- gated attention MIL
__global__ void attention_kernel(const float* __restrict__ h, const float* __restrict__ V,
                                 const float* __restrict__ U, const float* __restrict__ aw,
                                 float* __restrict__ z){
  __shared__ float sh[KCROPS*512];
  __shared__ float sc[KCROPS*128];
  __shared__ float at[KCROPS];
  const int n = blockIdx.x, tid = threadIdx.x;   // block of 128: tid = d
  for (int i = tid; i < KCROPS*512; i += blockDim.x) sh[i] = h[n*KCROPS*512 + i];
  __syncthreads();
  for (int k = 0; k < KCROPS; ++k) {
    float hv = 0.f, hu = 0.f;
    for (int j = 0; j < 512; ++j) { float x = sh[k*512 + j]; hv += x*V[j*128 + tid]; hu += x*U[j*128 + tid]; }
    sc[k*128 + tid] = tanhf(hv) * sigmoidf_(hu) * aw[tid];
  }
  __syncthreads();
  for (int off = 64; off; off >>= 1) {
    if (tid < off)
      for (int k = 0; k < KCROPS; ++k) sc[k*128 + tid] += sc[k*128 + tid + off];
    __syncthreads();
  }
  if (tid == 0) {
    float mx = -1e30f;
    for (int k = 0; k < KCROPS; ++k) mx = fmaxf(mx, sc[k*128]);
    float den = 0.f;
    for (int k = 0; k < KCROPS; ++k) { at[k] = __expf(sc[k*128] - mx); den += at[k]; }
    for (int k = 0; k < KCROPS; ++k) at[k] /= den;
  }
  __syncthreads();
  for (int d = tid; d < 512; d += blockDim.x) {
    float s = 0.f;
    for (int k = 0; k < KCROPS; ++k) s += at[k]*sh[k*512 + d];
    z[n*512 + d] = s;
  }
}

// ----------------------------------------------------- spatial max of h_g (NHWC)
__global__ void gvmax_kernel(const float* __restrict__ hg, float* __restrict__ gv){
  const int i = blockIdx.x*blockDim.x + threadIdx.x;  // (n*256 + c), 1024 total
  if (i >= N_IMG*256) return;
  const int n = i >> 8, c = i & 255;
  const float* p = hg + (size_t)n*CAMHW*256 + c;
  float mx = p[0];
  for (int j = 1; j < CAMHW; ++j) mx = fmaxf(mx, p[(size_t)j*256]);
  gv[i] = mx;
}

// ----------------------------------------------------- fusion GEMV + sigmoid
__global__ void fusion_kernel(const float* __restrict__ gv, const float* __restrict__ z,
                              const float* __restrict__ fw, const float* __restrict__ fb,
                              float* __restrict__ out){
  const int t = threadIdx.x;
  if (t >= N_IMG*3) return;
  const int n = t/3, c = t - 3*(t/3);
  float acc = fb[c];
  for (int i = 0; i < 256; ++i) acc += gv[n*256 + i]*fw[i*3 + c];
  for (int j = 0; j < 512; ++j) acc += z[n*512 + j]*fw[(256 + j)*3 + c];
  out[t] = sigmoidf_(acc);
}

// ============================================================================

static inline void conv_generic(hipStream_t s, const _Float16* in, const _Float16* w,
                                _Float16* oh, float* of,
                                int NB,int CI,int HI,int WI,int CO,int HO,int WO,
                                int ks,int act){
  long tiles = (long)((CO+15)/16) * (long)((NB*HO*WO+15)/16);
  int blocks = (int)((tiles + 7)/8);
  if (ks == 3) conv_wmma<3,2><<<blocks,256,0,s>>>(in,w,oh,of,NB,CI,HI,WI,CO,HO,WO,act);
  else         conv_wmma<1,1><<<blocks,256,0,s>>>(in,w,oh,of,NB,CI,HI,WI,CO,HO,WO,act);
}

static inline void conv3_blk(hipStream_t s, const _Float16* in, const _Float16* w,
                             _Float16* oh,
                             int NB,int CI,int HI,int WI,int CO,int HO,int WO){
  const int MT = (CI == 16) ? 2 : 4;     // CO=32 only for the CI=16 layer
  long tiles = (long)(CO/(MT*16)) * (long)((NB*HO*WO+31)/32);
  int blocks = (int)((tiles + 7)/8);
  switch (CI) {
    case 16:  conv3_wmma_blk<16,2> <<<blocks,256,0,s>>>(in,w,oh,NB,HI,WI,CO,HO,WO); break;
    case 32:  conv3_wmma_blk<32,4> <<<blocks,256,0,s>>>(in,w,oh,NB,HI,WI,CO,HO,WO); break;
    case 64:  conv3_wmma_blk<64,4> <<<blocks,256,0,s>>>(in,w,oh,NB,HI,WI,CO,HO,WO); break;
    case 128: conv3_wmma_blk<128,4><<<blocks,256,0,s>>>(in,w,oh,NB,HI,WI,CO,HO,WO); break;
    default:  conv3_wmma_blk<256,4><<<blocks,256,0,s>>>(in,w,oh,NB,HI,WI,CO,HO,WO); break;
  }
}

extern "C" void kernel_launch(void* const* d_in, const int* in_sizes, int n_in,
                              void* d_out, int out_size, void* d_ws, size_t ws_size,
                              hipStream_t stream) {
  (void)in_sizes; (void)n_in; (void)out_size; (void)ws_size;
  // ---- inputs (setup_inputs dict order, lists flattened) ----
  const float* x      = (const float*)d_in[0];
  const float* gW[6]  = { (const float*)d_in[1], (const float*)d_in[2], (const float*)d_in[3],
                          (const float*)d_in[4], (const float*)d_in[5], (const float*)d_in[6] };
  const float* postW  = (const float*)d_in[7];
  const float* salW   = (const float*)d_in[8];
  const float* lW[5]  = { (const float*)d_in[9],  (const float*)d_in[10], (const float*)d_in[11],
                          (const float*)d_in[12], (const float*)d_in[13] };
  const float* attV   = (const float*)d_in[14];
  const float* attU   = (const float*)d_in[15];
  const float* attw   = (const float*)d_in[16];
  // d_in[17]=cls_w, d_in[18]=cls_b : dead w.r.t. y_fusion
  const float* fuseW  = (const float*)d_in[19];
  const float* fuseB  = (const float*)d_in[20];
  float* out = (float*)d_out;

  // ---- workspace layout ----
  char* base = (char*)d_ws;
  size_t off = 0;
  auto alloc = [&](size_t bytes)->char*{
    char* p = base + off;
    off = (off + bytes + 255) & ~(size_t)255;
    return p;
  };
  _Float16* x16  = (_Float16*)alloc((size_t)4*2944*1920*2);        // 45.2 MB  (also L3/L5 out)
  _Float16* R1   = (_Float16*)alloc((size_t)4*16*1472*960*2);      // 180.9 MB (L1 out; local net later)
  _Float16* R2   = (_Float16*)alloc((size_t)4*32*736*480*2);       // 90.4 MB  (L2/L4/L6 out)
  const int gsz[6] = {144, 4608, 18432, 73728, 294912, 1179648};
  const int lsz[5] = {288, 18432, 73728, 294912, 1179648};
  _Float16* gW16[6]; for (int i=0;i<6;++i) gW16[i] = (_Float16*)alloc((size_t)gsz[i]*2);
  _Float16* postW16 = (_Float16*)alloc((size_t)131072*2);
  _Float16* salW16  = (_Float16*)alloc((size_t)768*2);
  _Float16* lW16[5]; for (int i=0;i<5;++i) lW16[i] = (_Float16*)alloc((size_t)lsz[i]*2);
  float*    hg32   = (float*)   alloc((size_t)4*256*CAMHW*4);
  _Float16* hg16   = (_Float16*)alloc((size_t)4*256*CAMHW*2);
  float*    salmap = (float*)   alloc((size_t)4*3*CAMHW*4);
  int*      pos    = (int*)     alloc(256);
  _Float16* crops  = (_Float16*)alloc((size_t)24*CROP*CROP*2);
  float*    hcrops = (float*)   alloc((size_t)24*512*4);
  float*    zbuf   = (float*)   alloc((size_t)4*512*4);
  float*    gv     = (float*)   alloc((size_t)4*256*4);
  // local-net ping-pong inside R1 (181 MB region)
  _Float16* lA = R1;
  _Float16* lB = (_Float16*)((char*)R1 + 33554432);

  // ---- convert input (CI=1: NHWC==NCHW); packed weights for generic convs;
  //      repacked [co][tap][ci] weights for blocked convs ----
  cvt_f32_f16<<<4096,256,0,stream>>>(x, x16, 4*2944*1920);
  cvt_f32_f16<<<1,256,0,stream>>>(gW[0], gW16[0], gsz[0]);           // g1: packed-K
  for (int i=1;i<6;++i)
    repack_w3<<<(gsz[i]+255)/256,256,0,stream>>>(gW[i], gW16[i], 16<<i, 8<<i);
  cvt_f32_f16<<<(131072+255)/256,256,0,stream>>>(postW, postW16, 131072);
  cvt_f32_f16<<<3,256,0,stream>>>(salW, salW16, 768);
  cvt_f32_f16<<<2,256,0,stream>>>(lW[0], lW16[0], lsz[0]);           // l1: packed-K
  for (int i=1;i<5;++i)
    repack_w3<<<(lsz[i]+255)/256,256,0,stream>>>(lW[i], lW16[i], 32<<i, 16<<i);

  // ---- global net: 6x (conv s2 3x3 + relu), NHWC ----
  conv_generic(stream, x16, gW16[0], R1, nullptr, 4, 1, 2944,1920, 16, 1472,960, 3, 1);
  conv3_blk(stream, R1,  gW16[1], R2,  4,  16, 1472, 960,  32,  736, 480);
  conv3_blk(stream, R2,  gW16[2], x16, 4,  32,  736, 480,  64,  368, 240);
  conv3_blk(stream, x16, gW16[3], R2,  4,  64,  368, 240, 128,  184, 120);
  conv3_blk(stream, R2,  gW16[4], x16, 4, 128,  184, 120, 256,   92,  60);
  conv3_blk(stream, x16, gW16[5], R2,  4, 256,   92,  60, 512,   46,  30);
  // post 1x1 + relu -> h_g (f16 for next conv, f32 for spatial max), NHWC
  conv_generic(stream, R2,  postW16, hg16, hg32,   4, 512,   46,  30, 256,   46,  30, 1, 1);
  // sal 1x1 + sigmoid -> saliency (f32, NHWC)
  conv_generic(stream, hg16, salW16, nullptr, salmap, 4, 256, 46, 30, 3, 46, 30, 1, 2);

  // ---- ROI retrieval + crops ----
  roi_kernel<<<4,256,0,stream>>>(salmap, pos);
  crops_kernel<<<dim3(256,24),256,0,stream>>>(x, pos, crops);

  // ---- local net: 5x (conv s2 3x3 + relu), NHWC ----
  conv_generic(stream, crops, lW16[0], lA, nullptr, 24, 1, 256,256, 32, 128,128, 3, 1);
  conv3_blk(stream, lA, lW16[1], lB, 24,  32, 128, 128,  64, 64, 64);
  conv3_blk(stream, lB, lW16[2], lA, 24,  64,  64,  64, 128, 32, 32);
  conv3_blk(stream, lA, lW16[3], lB, 24, 128,  32,  32, 256, 16, 16);
  conv3_blk(stream, lB, lW16[4], lA, 24, 256,  16,  16, 512,  8,  8);
  gap_kernel<<<48,256,0,stream>>>(lA, hcrops, 64);

  // ---- attention -> z ; global_vec ; fusion ----
  attention_kernel<<<4,128,0,stream>>>(hcrops, attV, attU, attw, zbuf);
  gvmax_kernel<<<4,256,0,stream>>>(hg32, gv);
  fusion_kernel<<<1,64,0,stream>>>(gv, zbuf, fuseW, fuseB, out);
}
